// BiLSTM_CRF_15418932592950
// MI455X (gfx1250) — matrix operand order repaired and checked by reading, not compiled, not run on hardware
//
#include <hip/hip_runtime.h>

// ---------------- problem constants ----------------
constexpr int S   = 4096;   // sequence length
constexpr int Lw  = 32;     // max word length
constexpr int CE  = 25;     // char emb dim
constexpr int CO  = 25;     // char-cnn out channels
constexpr int E   = 300;    // word emb dim
constexpr int KIN = E + CO; // 325
constexpr int KP  = 352;    // 325 padded to multiple of 32 (11 k-steps)
constexpr int H   = 256;    // lstm hidden
constexpr int G4  = 4 * H;  // 1024
constexpr int T   = 12;     // tagset
constexpr int TAG_START = 10;
constexpr int TAG_STOP  = 11;

// ---------------- types ----------------
typedef __bf16 bf16_t;
typedef __attribute__((ext_vector_type(16))) __bf16 v16bf;
typedef __attribute__((ext_vector_type(8)))  __bf16 v8bf;
typedef __attribute__((ext_vector_type(8)))  float  v8f;

__device__ __forceinline__ bf16_t f2bf(float f) {
    unsigned u = __builtin_bit_cast(unsigned, f);
    unsigned r = u + 0x7FFFu + ((u >> 16) & 1u);   // round-to-nearest-even
    unsigned short h = (unsigned short)(r >> 16);
    return __builtin_bit_cast(bf16_t, h);
}
__device__ __forceinline__ float sigmf(float x) { return 1.0f / (1.0f + __expf(-x)); }

// ---------------- kernel 1: char CNN + embedding gather -> Xp[S,KP] bf16 ----------------
__global__ void prep_embeds_k(const int* __restrict__ sentence,
                              const int* __restrict__ chars,
                              const float* __restrict__ word_embed,
                              const float* __restrict__ char_embed,
                              const float* __restrict__ conv_w,   // [CO,1,3,CE]
                              const float* __restrict__ conv_b,   // [CO]
                              bf16_t* __restrict__ Xp)            // [S,KP]
{
    __shared__ float ce[Lw][CE];
    __shared__ float cfeat[CO];
    const int s   = blockIdx.x;
    const int tid = threadIdx.x;         // blockDim = 128

    for (int idx = tid; idx < Lw * CE; idx += blockDim.x) {
        int p = idx / CE, c = idx - p * CE;
        int ch = chars[s * Lw + p];
        ce[p][c] = char_embed[ch * CE + c];
    }
    __syncthreads();

    if (tid < CO) {
        float mx = -1e30f;
        for (int p = 0; p < Lw + 2; ++p) {          // out length = 34
            float acc = conv_b[tid];
            #pragma unroll
            for (int kk = 0; kk < 3; ++kk) {
                int q = p + kk - 2;
                if (q >= 0 && q < Lw) {
                    const float* wv = conv_w + tid * (3 * CE) + kk * CE;
                    for (int c = 0; c < CE; ++c) acc += ce[q][c] * wv[c];
                }
            }
            mx = fmaxf(mx, acc);
        }
        cfeat[tid] = mx;
    }

    const int w = sentence[s];
    for (int k = tid; k < E; k += blockDim.x)
        Xp[(size_t)s * KP + k] = f2bf(word_embed[(size_t)w * E + k]);
    __syncthreads();
    for (int k = E + tid; k < KP; k += blockDim.x)
        Xp[(size_t)s * KP + k] = f2bf(k < KIN ? cfeat[k - E] : 0.0f);
}

// ---------------- kernel 2: fp32 -> bf16 weight conversion (with K padding) ----------------
__global__ void cvt_pad_bf16_k(const float* __restrict__ W, bf16_t* __restrict__ Wp,
                               int K, int KPad, int total)
{
    for (int idx = blockIdx.x * blockDim.x + threadIdx.x; idx < total;
         idx += gridDim.x * blockDim.x) {
        int n = idx / KPad, k = idx - n * KPad;
        float v = (k < K) ? W[(size_t)n * K + k] : 0.0f;
        Wp[idx] = f2bf(v);
    }
}

// ---------------- kernel 3: input GEMM via WMMA bf16: G[S,1024] = Xp @ W.T + b ----------------
__global__ void input_gemm_wmma_k(const bf16_t* __restrict__ A,   // [S,KP]
                                  const bf16_t* __restrict__ W,   // [G4,KP] row n = weights of output n
                                  const float*  __restrict__ bias,// [G4]
                                  float* __restrict__ G)          // [S,G4]
{
    const int lane = threadIdx.x & 31;
    const int wave = threadIdx.x >> 5;                 // 4 waves/block
    const int mt   = blockIdx.x;                       // S/16 tiles
    const int nt   = blockIdx.y * 4 + wave;            // G4/16 tiles
    const int m0   = mt * 16, n0 = nt * 16;
    const int hi   = lane >> 4;                        // 0 or 1
    const int ml   = lane & 15;

    const bf16_t* arow = A + (size_t)(m0 + ml) * KP;
    const bf16_t* wrow = W + (size_t)(n0 + ml) * KP;

    v8f acc = {};
    for (int kt = 0; kt < KP / 32; ++kt) {
        const int k0 = kt * 32;
        // A fragment: halves 0..7 = K[k0+hi*8 ..], halves 8..15 = K[k0+16+hi*8 ..]
        v8bf a0 = *(const v8bf*)(arow + k0 + hi * 8);
        v8bf a1 = *(const v8bf*)(arow + k0 + 16 + hi * 8);
        // B fragment: halves 0..7 = K[k0+hi*8 ..], halves 8..15 = K[k0+16+hi*8 ..] of column ml
        v8bf b0 = *(const v8bf*)(wrow + k0 + hi * 8);
        v8bf b1 = *(const v8bf*)(wrow + k0 + 16 + hi * 8);
        v16bf af, bfr;
        #pragma unroll
        for (int i = 0; i < 8; ++i) {
            af[i] = a0[i];  af[8 + i] = a1[i];
            bfr[i] = b0[i]; bfr[8 + i] = b1[i];
        }
        acc = __builtin_amdgcn_wmma_f32_16x16x32_bf16(
                  false, af, false, bfr, (short)0, acc, false, false);
    }
    const float bn = bias[n0 + ml];
    float* gout = G + (size_t)(m0 + hi * 8) * G4 + n0 + ml;
    #pragma unroll
    for (int r = 0; r < 8; ++r) gout[(size_t)r * G4] = acc[r] + bn;
}

// ---------------- kernel 4: persistent sequential LSTM via WMMA matvec ----------------
// One 1024-thread block per direction (32 waves). Wave w owns gate rows [32w, 32w+32):
// two 16x16 M-tiles, K=256 -> 16 v_wmma per step. h is the (broadcast) B operand from LDS.
// The per-step input-projection row G[s,0:1024] is double-buffer staged into LDS with the
// CDNA5 async-to-LDS path (ASYNCcnt), overlapping its latency with the WMMA matvec.
__global__ void __launch_bounds__(1024)
lstm_seq_wmma_k(const float* __restrict__ Gf, const float* __restrict__ Gb,
                const bf16_t* __restrict__ Whf, const bf16_t* __restrict__ Whb,
                float* __restrict__ hsf, float* __restrict__ hsb)
{
    const int dir = blockIdx.x;
    const float*  Gin = dir ? Gb : Gf;
    const bf16_t* Whh = dir ? Whb : Whf;
    float*        hs  = dir ? hsb : hsf;

    __shared__ __align__(16) bf16_t hbf[H];     // h (bf16) = B operand source
    __shared__ float csh[H];                    // cell state
    __shared__ float gsh[G4];                   // Whh @ h result
    __shared__ float gq[2][G4];                 // double-buffered input-projection row

    const int j    = threadIdx.x;               // 1024 threads
    const int lane = j & 31;
    const int wv   = j >> 5;                    // wave id 0..31
    const int hi   = lane >> 4;
    const int ml   = lane & 15;

    if (j < H) { hbf[j] = f2bf(0.0f); csh[j] = 0.0f; }

    // preload Gin row for t = 0 into gq[0]
    {
        const int s0 = dir ? (S - 1) : 0;
        unsigned ldsa = (unsigned)(uintptr_t)&gq[0][j];
        unsigned voff = (unsigned)((s0 * G4 + j) * 4);
        asm volatile("global_load_async_to_lds_b32 %0, %1, %2"
                     :: "v"(ldsa), "v"(voff), "s"((unsigned long long)Gin) : "memory");
        asm volatile("s_wait_asynccnt 0x0" ::: "memory");
    }
    __syncthreads();

    const bf16_t* wrow0 = Whh + (size_t)(32 * wv + ml) * H;       // M-tile 0 rows
    const bf16_t* wrow1 = Whh + (size_t)(32 * wv + 16 + ml) * H;  // M-tile 1 rows

    for (int t = 0; t < S; ++t) {
        const int cur = t & 1;
        // stage next step's input-projection row asynchronously into the other buffer
        if (t + 1 < S) {
            const int sn = dir ? (S - 2 - t) : (t + 1);
            unsigned ldsa = (unsigned)(uintptr_t)&gq[cur ^ 1][j];
            unsigned voff = (unsigned)((sn * G4 + j) * 4);
            asm volatile("global_load_async_to_lds_b32 %0, %1, %2"
                         :: "v"(ldsa), "v"(voff), "s"((unsigned long long)Gin) : "memory");
        }
        // B fragments: h K-slices, broadcast across lanes from LDS
        v16bf bfrag[8];
        #pragma unroll
        for (int kt = 0; kt < 8; ++kt) {
            v8bf h0 = *(const v8bf*)(hbf + kt * 32 + hi * 8);
            v8bf h1 = *(const v8bf*)(hbf + kt * 32 + 16 + hi * 8);
            #pragma unroll
            for (int i = 0; i < 8; ++i) { bfrag[kt][i] = h0[i]; bfrag[kt][8 + i] = h1[i]; }
        }
        // two 16-row tiles of Whh @ h via WMMA (weights = A operand, L0-resident)
        #pragma unroll
        for (int mtl = 0; mtl < 2; ++mtl) {
            const bf16_t* wr = mtl ? wrow1 : wrow0;
            v8f acc = {};
            #pragma unroll
            for (int kt = 0; kt < 8; ++kt) {
                const int k0 = kt * 32;
                v8bf a0 = *(const v8bf*)(wr + k0 + hi * 8);
                v8bf a1 = *(const v8bf*)(wr + k0 + 16 + hi * 8);
                v16bf af;
                #pragma unroll
                for (int i = 0; i < 8; ++i) { af[i] = a0[i]; af[8 + i] = a1[i]; }
                acc = __builtin_amdgcn_wmma_f32_16x16x32_bf16(
                          false, af, false, bfrag[kt], (short)0, acc, false, false);
            }
            if (ml == 0) {      // column 0 of D holds the matvec result
                #pragma unroll
                for (int r = 0; r < 8; ++r)
                    gsh[32 * wv + mtl * 16 + hi * 8 + r] = acc[r];
            }
        }
        // current buffer is the older outstanding async load -> wait count <= 1
        asm volatile("s_wait_asynccnt 0x1" ::: "memory");
        __syncthreads();
        // gate nonlinearities + state update
        if (j < H) {
            const int s = dir ? (S - 1 - t) : t;
            float ig = sigmf(gq[cur][j]         + gsh[j]);
            float fg = sigmf(gq[cur][H + j]     + gsh[H + j]);
            float gg = tanhf(gq[cur][2 * H + j] + gsh[2 * H + j]);
            float og = sigmf(gq[cur][3 * H + j] + gsh[3 * H + j]);
            float c  = fg * csh[j] + ig * gg;
            float h  = og * tanhf(c);
            csh[j] = c;
            hbf[j] = f2bf(h);
            hs[(size_t)s * H + j] = h;
        }
        __syncthreads();
    }
}

// ---------------- kernel 5: tag projection feats[S,T] ----------------
__global__ void tag_proj_k(const float* __restrict__ hsf, const float* __restrict__ hsb,
                           const float* __restrict__ Wtag, const float* __restrict__ btag,
                           float* __restrict__ feats)
{
    const int idx = blockIdx.x * blockDim.x + threadIdx.x;
    if (idx >= S * T) return;
    const int s = idx / T, t = idx - s * T;
    const float* wf = Wtag + (size_t)t * (2 * H);
    const float* hf = hsf + (size_t)s * H;
    const float* hb = hsb + (size_t)s * H;
    float acc = btag[t];
    for (int k = 0; k < H; ++k) acc += hf[k] * wf[k];
    for (int k = 0; k < H; ++k) acc += hb[k] * wf[H + k];
    feats[idx] = acc;
}

// ---------------- kernel 6: CRF forward + gold score, one wave32 ----------------
__global__ void crf_forward_k(const float* __restrict__ feats,  // [S,T]
                              const float* __restrict__ trans,  // [T,T]
                              const int*   __restrict__ tags,   // [S]
                              float* __restrict__ out)
{
    const int lane = threadIdx.x;          // 32 threads, 1 wave
    float tcol[T];
    #pragma unroll
    for (int i = 0; i < T; ++i)
        tcol[i] = (lane < T) ? trans[i * T + lane] : 0.0f;

    float alpha = (lane == TAG_START) ? 0.0f : -10000.0f;
    for (int s = 0; s < S; ++s) {
        const float f = (lane < T) ? feats[s * T + lane] : 0.0f;
        float av[T];
        float m = -1e30f;
        #pragma unroll
        for (int i = 0; i < T; ++i) {
            av[i] = __shfl(alpha, i, 32) + tcol[i];
            m = fmaxf(m, av[i]);
        }
        float sum = 0.0f;
        #pragma unroll
        for (int i = 0; i < T; ++i) sum += __expf(av[i] - m);
        const float na = f + m + __logf(sum);
        alpha = (lane < T) ? na : -10000.0f;
    }

    // forward_score = logsumexp(alpha + trans[:, STOP])
    const float v = (lane < T) ? (alpha + trans[lane * T + TAG_STOP]) : -1e30f;
    float m = -1e30f;
    #pragma unroll
    for (int i = 0; i < T; ++i) m = fmaxf(m, __shfl(v, i, 32));
    float sum = 0.0f;
    #pragma unroll
    for (int i = 0; i < T; ++i) sum += __expf(__shfl(v, i, 32) - m);
    const float fwd = m + __logf(sum);

    // gold path score (strided over lanes, wave reduce)
    float gold = 0.0f;
    for (int s = lane; s < S; s += 32) {
        const int tg   = tags[s];
        const int prev = (s == 0) ? TAG_START : tags[s - 1];
        gold += feats[s * T + tg] + trans[prev * T + tg];
    }
    #pragma unroll
    for (int off = 16; off > 0; off >>= 1) gold += __shfl_down(gold, off, 32);
    if (lane == 0) {
        gold += trans[tags[S - 1] * T + TAG_STOP];
        out[0] = fwd - gold;
    }
}

// ---------------- host launch ----------------
extern "C" void kernel_launch(void* const* d_in, const int* in_sizes, int n_in,
                              void* d_out, int out_size, void* d_ws, size_t ws_size,
                              hipStream_t stream) {
    const int*   sentence   = (const int*)  d_in[0];
    const int*   chars      = (const int*)  d_in[1];
    const int*   tags       = (const int*)  d_in[2];
    /* caps (d_in[3]) unused by reference */
    const float* word_embed = (const float*)d_in[4];
    const float* char_embed = (const float*)d_in[5];
    const float* conv_w     = (const float*)d_in[6];
    const float* conv_b     = (const float*)d_in[7];
    const float* Wih_f      = (const float*)d_in[8];
    const float* Whh_f      = (const float*)d_in[9];
    const float* b_f        = (const float*)d_in[10];
    const float* Wih_b      = (const float*)d_in[11];
    const float* Whh_b      = (const float*)d_in[12];
    const float* b_b        = (const float*)d_in[13];
    const float* W_tag      = (const float*)d_in[14];
    const float* b_tag      = (const float*)d_in[15];
    const float* trans      = (const float*)d_in[16];

    char* p = (char*)d_ws;
    auto alloc = [&](size_t bytes) {
        char* r = p;
        p += (bytes + 255) & ~(size_t)255;
        return r;
    };
    bf16_t* Xp    = (bf16_t*)alloc((size_t)S * KP * sizeof(bf16_t));
    bf16_t* Wfp   = (bf16_t*)alloc((size_t)G4 * KP * sizeof(bf16_t));
    bf16_t* Wbp   = (bf16_t*)alloc((size_t)G4 * KP * sizeof(bf16_t));
    bf16_t* Whhf  = (bf16_t*)alloc((size_t)G4 * H * sizeof(bf16_t));
    bf16_t* Whhb  = (bf16_t*)alloc((size_t)G4 * H * sizeof(bf16_t));
    float*  Gf    = (float*) alloc((size_t)S * G4 * sizeof(float));
    float*  Gb    = (float*) alloc((size_t)S * G4 * sizeof(float));
    float*  hsf   = (float*) alloc((size_t)S * H * sizeof(float));
    float*  hsb   = (float*) alloc((size_t)S * H * sizeof(float));
    float*  feats = (float*) alloc((size_t)S * T * sizeof(float));

    // 1. char CNN + word embedding gather, bf16 pack
    prep_embeds_k<<<S, 128, 0, stream>>>(sentence, chars, word_embed, char_embed,
                                         conv_w, conv_b, Xp);
    // 2. weight conversion to bf16 (K-padded)
    {
        int tot1 = G4 * KP;
        cvt_pad_bf16_k<<<(tot1 + 255) / 256, 256, 0, stream>>>(Wih_f, Wfp, KIN, KP, tot1);
        cvt_pad_bf16_k<<<(tot1 + 255) / 256, 256, 0, stream>>>(Wih_b, Wbp, KIN, KP, tot1);
        int tot2 = G4 * H;
        cvt_pad_bf16_k<<<(tot2 + 255) / 256, 256, 0, stream>>>(Whh_f, Whhf, H, H, tot2);
        cvt_pad_bf16_k<<<(tot2 + 255) / 256, 256, 0, stream>>>(Whh_b, Whhb, H, H, tot2);
    }
    // 3. input GEMMs via WMMA: G = Xp @ Wih.T + b  (256 M-tiles x 64 N-tiles)
    input_gemm_wmma_k<<<dim3(S / 16, G4 / 16 / 4), 128, 0, stream>>>(Xp, Wfp, b_f, Gf);
    input_gemm_wmma_k<<<dim3(S / 16, G4 / 16 / 4), 128, 0, stream>>>(Xp, Wbp, b_b, Gb);
    // 4. sequential BiLSTM: WMMA matvec recurrence, 2 persistent workgroups
    lstm_seq_wmma_k<<<2, 1024, 0, stream>>>(Gf, Gb, Whhf, Whhb, hsf, hsb);
    // 5. tag projection
    tag_proj_k<<<(S * T + 255) / 256, 256, 0, stream>>>(hsf, hsb, W_tag, b_tag, feats);
    // 6. CRF forward + gold, scalar out
    crf_forward_k<<<1, 32, 0, stream>>>(feats, trans, tags, (float*)d_out);
}
